// SelfAttention_6511170421217
// MI455X (gfx1250) — matrix operand order repaired and checked by reading, not compiled, not run on hardware
//
#include <hip/hip_runtime.h>

#define NHEAD 16
#define DHEAD 64
#define SEQ   1024
#define QTILE 128   // query rows per block
#define KTILE 64    // keys per iteration
#define WAVES 8

typedef _Float16 v16h __attribute__((ext_vector_type(16)));
typedef _Float16 v8h  __attribute__((ext_vector_type(8)));
typedef _Float16 v4h  __attribute__((ext_vector_type(4)));
typedef float    v8f  __attribute__((ext_vector_type(8)));

static __device__ __forceinline__ v16h pack16(const _Float16* p0, const _Float16* p1) {
  v8h a = *(const v8h*)p0;
  v8h b = *(const v8h*)p1;
  v16h r;
#pragma unroll
  for (int i = 0; i < 8; ++i) { r[i] = a[i]; r[i + 8] = b[i]; }
  return r;
}

static __device__ __forceinline__ float rmax16(float x) {
#pragma unroll
  for (int m = 1; m < 16; m <<= 1) x = fmaxf(x, __shfl_xor(x, m, 32));
  return x;
}

#define WMMA_F16(A, B, C) \
  __builtin_amdgcn_wmma_f32_16x16x32_f16(false, (A), false, (B), (short)0, (C), false, false)

__global__ __launch_bounds__(256, 2)
void attn_fwd(const float* __restrict__ x, float* __restrict__ out) {
  __shared__ _Float16 ldsQ[QTILE * DHEAD];          // 16 KB, q tile (pre-scaled by 0.125*log2e)
  __shared__ _Float16 ldsK[KTILE * DHEAD];          // 8 KB, key tile row-major [key][d]
  __shared__ _Float16 ldsVt[DHEAD * KTILE];         // 8 KB, value tile transposed [d][key]
  __shared__ _Float16 ldsP[WAVES][16 * KTILE];      // 16 KB, per-wave P scratch [row][key]

  const int tid  = threadIdx.x;
  const int lane = tid & 31;
  const int wave = tid >> 5;
  const int ln   = lane & 15;   // N / row-in-16 index
  const int g    = lane >> 4;   // half-wave group

  const int qblk = blockIdx.x * QTILE;
  const int h    = blockIdx.y & (NHEAD - 1);
  const int b    = blockIdx.y >> 4;

  const size_t rowstride = 3 * NHEAD * DHEAD;       // 3072 floats per (b,s)
  const float* xb = x + (size_t)b * SEQ * rowstride;
  const float* qg = xb + (size_t)h * DHEAD;
  const float* kg = xb + (size_t)(NHEAD * DHEAD) + (size_t)h * DHEAD;
  const float* vg = xb + (size_t)(2 * NHEAD * DHEAD) + (size_t)h * DHEAD;

  // scale = 1/sqrt(64) * log2(e): base-2 softmax (v_exp_f32 is exp2)
  const float kscale = 0.125f * 1.44269504088896f;

  // ---- per-thread staging geometry (loop-invariant) ----
  const int r0 = tid >> 4;          // staging row 0..15
  const int d0 = (tid & 15) << 2;   // staging col (float4 granular)

  // ---- stage Q tile (float4 loads -> packed f16 stores) ----
  {
    const float* qptr = qg + (size_t)(qblk + r0) * rowstride + d0;
    _Float16* lq = &ldsQ[r0 * DHEAD + d0];
#pragma unroll
    for (int i = 0; i < QTILE / 16; ++i) {
      float4 q = *(const float4*)(qptr + (size_t)(i * 16) * rowstride);
      v4h qh = { (_Float16)(q.x * kscale), (_Float16)(q.y * kscale),
                 (_Float16)(q.z * kscale), (_Float16)(q.w * kscale) };
      *(v4h*)(lq + i * 16 * DHEAD) = qh;
    }
  }
  __syncthreads();

  // per-wave Q A-fragments: d-chunks [0,32) and [32,64)
  const _Float16* qp = &ldsQ[(wave * 16 + ln) * DHEAD];
  v16h qa0 = pack16(qp + 0  + g * 16, qp + 8  + g * 16);
  v16h qa1 = pack16(qp + 32 + g * 16, qp + 40 + g * 16);

  v16h onesB;
#pragma unroll
  for (int i = 0; i < 16; ++i) onesB[i] = (_Float16)1.0f;

  // loop-invariant LDS fragment pointers
  const _Float16* k0 = &ldsK[(0 * 16 + ln) * DHEAD + g * 16];
  const _Float16* k1 = &ldsK[(1 * 16 + ln) * DHEAD + g * 16];
  const _Float16* k2 = &ldsK[(2 * 16 + ln) * DHEAD + g * 16];
  const _Float16* k3 = &ldsK[(3 * 16 + ln) * DHEAD + g * 16];
  const _Float16* vp0 = &ldsVt[(0 * 16 + ln) * KTILE + g * 16];
  const _Float16* vp1 = &ldsVt[(1 * 16 + ln) * KTILE + g * 16];
  const _Float16* vp2 = &ldsVt[(2 * 16 + ln) * KTILE + g * 16];
  const _Float16* vp3 = &ldsVt[(3 * 16 + ln) * KTILE + g * 16];
  const _Float16* pp  = &ldsP[wave][ln * KTILE];
  _Float16* pw = &ldsP[wave][(8 * g) * KTILE + ln];

  // per-tile-advanced staging pointers
  const float* kptr = kg + (size_t)r0 * rowstride + d0;
  const float* vptr = vg + (size_t)r0 * rowstride + d0;
  _Float16* lK  = &ldsK[r0 * DHEAD + d0];
  _Float16* lV0 = &ldsVt[(d0 + 0) * KTILE + r0];
  _Float16* lV1 = &ldsVt[(d0 + 1) * KTILE + r0];
  _Float16* lV2 = &ldsVt[(d0 + 2) * KTILE + r0];
  _Float16* lV3 = &ldsVt[(d0 + 3) * KTILE + r0];

  v8f o0 = {}, o1 = {}, o2 = {}, o3 = {};
  v8f lacc = {};
  float mrow[8];
#pragma unroll
  for (int v = 0; v < 8; ++v) mrow[v] = -1e30f;

  // wave-uniform scalar copies so branches are s_cbranch (EXEC untouched; WMMA
  // requires EXEC == all-ones)
  const int qw  = __builtin_amdgcn_readfirstlane(qblk + wave * 16);
  const int nkt = qblk / KTILE + QTILE / KTILE;     // causal bound for the block

  for (int kt = 0; kt < nkt; ++kt) {
    const int kb = kt * KTILE;
    __syncthreads();                                // WAR on K/V LDS tiles
    // ---- stage K (row-major) and V (transposed) tiles ----
#pragma unroll
    for (int i = 0; i < KTILE / 16; ++i) {
      const size_t go = (size_t)(i * 16) * rowstride;
      float4 kv = *(const float4*)(kptr + go);
      v4h kh = { (_Float16)kv.x, (_Float16)kv.y, (_Float16)kv.z, (_Float16)kv.w };
      *(v4h*)(lK + i * 16 * DHEAD) = kh;
      float4 vv = *(const float4*)(vptr + go);
      lV0[i * 16] = (_Float16)vv.x;
      lV1[i * 16] = (_Float16)vv.y;
      lV2[i * 16] = (_Float16)vv.z;
      lV3[i * 16] = (_Float16)vv.w;
    }
    kptr += (size_t)KTILE * rowstride;
    vptr += (size_t)KTILE * rowstride;
    if (kt + 1 < nkt) {                             // prefetch next tile (speculative)
      __builtin_prefetch(kptr, 0, 0);
      __builtin_prefetch(kptr + 32 * rowstride, 0, 0);
      __builtin_prefetch(vptr, 0, 0);
      __builtin_prefetch(vptr + 32 * rowstride, 0, 0);
    }
    __syncthreads();

    // fully-masked tile for this wave? (scalar condition -> s_cbranch)
    if (kb > qw + 15) continue;

    // ---- S = Q * K^T : 16 x 64 scores, four 16-key N tiles ----
    v8f s0 = {}, s1 = {}, s2 = {}, s3 = {};
    {
      v16h b0 = pack16(k0 + 0, k0 + 8);
      v16h b1 = pack16(k1 + 0, k1 + 8);
      v16h b2 = pack16(k2 + 0, k2 + 8);
      v16h b3 = pack16(k3 + 0, k3 + 8);
      s0 = WMMA_F16(qa0, b0, s0);
      s1 = WMMA_F16(qa0, b1, s1);
      s2 = WMMA_F16(qa0, b2, s2);
      s3 = WMMA_F16(qa0, b3, s3);
      b0 = pack16(k0 + 32, k0 + 40);
      b1 = pack16(k1 + 32, k1 + 40);
      b2 = pack16(k2 + 32, k2 + 40);
      b3 = pack16(k3 + 32, k3 + 40);
      s0 = WMMA_F16(qa1, b0, s0);
      s1 = WMMA_F16(qa1, b1, s1);
      s2 = WMMA_F16(qa1, b2, s2);
      s3 = WMMA_F16(qa1, b3, s3);
    }

    // ---- online softmax over this key tile (base-2) ----
    const int colbase = kb + ln;

#define SOFTMAX_LOOP(MASKED)                                                   \
  _Pragma("unroll")                                                            \
  for (int v = 0; v < 8; ++v) {                                                \
    const int row = qw + v + 8 * g;                                            \
    float e0 = s0[v], e1 = s1[v], e2 = s2[v], e3 = s3[v];                      \
    if (MASKED) {                                                              \
      e0 = (colbase      <= row) ? e0 : -1e9f;                                 \
      e1 = (colbase + 16 <= row) ? e1 : -1e9f;                                 \
      e2 = (colbase + 32 <= row) ? e2 : -1e9f;                                 \
      e3 = (colbase + 48 <= row) ? e3 : -1e9f;                                 \
    }                                                                          \
    float mt = rmax16(fmaxf(fmaxf(e0, e1), fmaxf(e2, e3)));                    \
    float mn = fmaxf(mrow[v], mt);                                             \
    float alpha = __builtin_amdgcn_exp2f(mrow[v] - mn);                        \
    mrow[v] = mn;                                                              \
    float p0 = __builtin_amdgcn_exp2f(e0 - mn);                                \
    float p1 = __builtin_amdgcn_exp2f(e1 - mn);                                \
    float p2 = __builtin_amdgcn_exp2f(e2 - mn);                                \
    float p3 = __builtin_amdgcn_exp2f(e3 - mn);                                \
    _Float16* prw = pw + v * KTILE;                                            \
    prw[0]  = (_Float16)p0; prw[16] = (_Float16)p1;                            \
    prw[32] = (_Float16)p2; prw[48] = (_Float16)p3;                            \
    lacc[v] *= alpha;                                                          \
    o0[v] *= alpha; o1[v] *= alpha; o2[v] *= alpha; o3[v] *= alpha;            \
  }

    if (kb + KTILE - 1 > qw) {                      // diagonal tile: mask needed
      SOFTMAX_LOOP(true)
    } else {                                        // interior tile: no mask
      SOFTMAX_LOOP(false)
    }
#undef SOFTMAX_LOOP

    // wave-local LDS RAW fence for the P scratch (LDS ops within a wave are
    // in-order, so WAR against next iteration's stores is safe)
    asm volatile("s_wait_dscnt 0" ::: "memory");

    // ---- P A-fragments (keys [0,32) and [32,64)) ----
    v16h pa0 = pack16(pp + 0  + g * 16, pp + 8  + g * 16);
    v16h pa1 = pack16(pp + 32 + g * 16, pp + 40 + g * 16);

    // row sums on the matrix unit: l += P * ones
    lacc = WMMA_F16(pa0, onesB, lacc);
    lacc = WMMA_F16(pa1, onesB, lacc);

    // ---- O += P * V ----
    {
      v16h b0 = pack16(vp0, vp0 + 8);
      v16h b1 = pack16(vp1, vp1 + 8);
      v16h b2 = pack16(vp2, vp2 + 8);
      v16h b3 = pack16(vp3, vp3 + 8);
      o0 = WMMA_F16(pa0, b0, o0);
      o1 = WMMA_F16(pa0, b1, o1);
      o2 = WMMA_F16(pa0, b2, o2);
      o3 = WMMA_F16(pa0, b3, o3);
      b0 = pack16(vp0 + 32, vp0 + 40);
      b1 = pack16(vp1 + 32, vp1 + 40);
      b2 = pack16(vp2 + 32, vp2 + 40);
      b3 = pack16(vp3 + 32, vp3 + 40);
      o0 = WMMA_F16(pa1, b0, o0);
      o1 = WMMA_F16(pa1, b1, o1);
      o2 = WMMA_F16(pa1, b2, o2);
      o3 = WMMA_F16(pa1, b3, o3);
    }
  }

  // ---- normalize and write out[b][s][h*64 + d] ----
  float* ob = out + ((size_t)b * SEQ) * (NHEAD * DHEAD) + (size_t)h * DHEAD
            + (size_t)(qw + 8 * g) * (NHEAD * DHEAD) + ln;
#pragma unroll
  for (int v = 0; v < 8; ++v) {
    float inv = 1.0f / lacc[v];
    float* orow = ob + (size_t)v * (NHEAD * DHEAD);
    orow[0]  = o0[v] * inv;
    orow[16] = o1[v] * inv;
    orow[32] = o2[v] * inv;
    orow[48] = o3[v] * inv;
  }
}

extern "C" void kernel_launch(void* const* d_in, const int* in_sizes, int n_in,
                              void* d_out, int out_size, void* d_ws, size_t ws_size,
                              hipStream_t stream) {
  (void)n_in; (void)d_ws; (void)ws_size; (void)out_size;
  const float* x = (const float*)d_in[0];   // (B, S, 3*1024) fp32; d_in[1] mask unused (causal hardcoded)
  float* out = (float*)d_out;               // (B, S, 1024) fp32
  const int B = in_sizes[0] / (SEQ * 3 * NHEAD * DHEAD);
  dim3 grid(SEQ / QTILE, B * NHEAD);
  attn_fwd<<<grid, 256, 0, stream>>>(x, out);
}
